// ConvLayer_38740605010103
// MI455X (gfx1250) — matrix operand order repaired and checked by reading, not compiled, not run on hardware
//
#include <hip/hip_runtime.h>
#include <hip/hip_bf16.h>

#define NA 100000          // atoms
#define MN 12              // neighbors per atom
#define FA 64              // atom_fea_len
#define KD 256             // 2F + NBR
#define CD 128             // 2F
#define R1 1200000.0f      // rows for BN1 stats = NA*MN
#define R2 100000.0f       // rows for BN2 stats = NA
#define EPSV 1e-3f

typedef __attribute__((ext_vector_type(16))) __bf16 v16bf;
typedef __attribute__((ext_vector_type(8)))  float  v8f;

union FragB16 { v16bf v; unsigned u[8]; };
union FragF32 { v8f  v; float   f[8]; };

__device__ __forceinline__ unsigned pack_bf16(float a, float b) {
    unsigned ua = __float_as_uint(a), ub = __float_as_uint(b);
    ua = (ua + 0x7FFFu + ((ua >> 16) & 1u)) >> 16;   // RNE f32->bf16
    ub = (ub + 0x7FFFu + ((ub >> 16) & 1u)) >> 16;
    return ua | (ub << 16);
}

__device__ __forceinline__ float softplusf(float x) {
    return x > 20.f ? x : log1pf(__expf(x));
}

// ---------------------------------------------------------------------------
// Kernel 0: zero BN-stat accumulators; convert W (f32 [256][128]) into bf16
// words pre-swizzled into this kernel's WMMA B-fragment order (contiguous k, as in A):
//   word t = ((kk*8 + n)*32 + L)*8 + i  holds K pair (k0, k0+1) for
//   col = 16n + (L&15), k0 = 32kk + (L>=16 ? 16 : 0) + 2i.
// ---------------------------------------------------------------------------
__global__ __launch_bounds__(256) void conv_prep(const float* __restrict__ W,
                                                 unsigned* __restrict__ Wb,
                                                 float* __restrict__ stats) {
    int t = blockIdx.x * 256 + threadIdx.x;
    if (t < 384) stats[t] = 0.f;           // stats1[256] ++ stats2[128]
    if (t < 16384) {
        int i   = t & 7;
        int L   = (t >> 3) & 31;
        int rest = t >> 8;                 // kk*8 + n
        int kk  = rest >> 3;
        int n   = rest & 7;
        int col = n * 16 + (L & 15);
        int k0  = kk * 32 + ((L >= 16) ? 16 : 0) + 2 * i;
        Wb[t] = pack_bf16(W[k0 * CD + col], W[(k0 + 1) * CD + col]);
    }
}

// ---------------------------------------------------------------------------
// GEMM kernels. One atom per wave: 16-row A tile (rows 12..15 are zero pad,
// masked out of all reductions). 8 waves / block -> 8 atoms / block.
// PASS==1: accumulate per-column sum / sumsq of total_gated (BN1 stats).
// PASS==2: BN1 + sigmoid*softplus gate + neighbor-sum -> nbr_sumed, BN2 stats.
// ---------------------------------------------------------------------------
template <int PASS>
__global__ __launch_bounds__(256) void conv_gemm(
    const float* __restrict__ atom, const float* __restrict__ nbrf,
    const int* __restrict__ idx, const unsigned* __restrict__ Wb,
    const float* __restrict__ gamma1, const float* __restrict__ beta1,
    float* __restrict__ stats1, float* __restrict__ stats2,
    float* __restrict__ nbr_sumed) {
    __shared__ unsigned ldsW[8192];     // 32KB: half of W's bf16 fragments
    __shared__ float    ldsStat[256];

    const int tid  = threadIdx.x;
    const int wave = tid >> 5;
    const int lane = tid & 31;
    const int mrow = lane & 15;
    const bool hi  = lane >= 16;

    ldsStat[tid] = 0.f;

    const int a = blockIdx.x * 8 + wave;
    const bool active = a < NA;

    // ---- Build the 8 A fragments (16x32 bf16 tiles) in registers ----------
    FragB16 afr[8];
#pragma unroll
    for (int kk = 0; kk < 8; ++kk)
#pragma unroll
        for (int i = 0; i < 8; ++i) afr[kk].u[i] = 0u;

    if (active) {
        const bool rowvalid = mrow < MN;
        const float* selfrow = atom + (long)a * FA;
        const float* nbrow   = selfrow;    // safe dummy; overwritten if valid
        const float* edgerow = selfrow;
        if (rowvalid) {
            int nb  = idx[a * MN + mrow];
            nbrow   = atom + (long)nb * FA;
            edgerow = nbrf + ((long)a * MN + mrow) * 128;
        }
#pragma unroll
        for (int kk = 0; kk < 8; ++kk) {
            // concat regions are compile-time per kk: 0,1->self 2,3->nbr 4..7->edge
            const float* base = (kk < 2) ? selfrow : (kk < 4 ? nbrow : edgerow);
            const int roff    = (kk < 2) ? 0 : (kk < 4 ? 64 : 128);
#pragma unroll
            for (int c = 0; c < 2; ++c) {
                int s = kk * 32 + (hi ? 16 : 0) + c * 8 - roff;
                float4 x = make_float4(0.f, 0.f, 0.f, 0.f), y = x;
                if (rowvalid) {
                    x = *(const float4*)(base + s);
                    y = *(const float4*)(base + s + 4);
                }
                afr[kk].u[c * 4 + 0] = pack_bf16(x.x, x.y);
                afr[kk].u[c * 4 + 1] = pack_bf16(x.z, x.w);
                afr[kk].u[c * 4 + 2] = pack_bf16(y.x, y.y);
                afr[kk].u[c * 4 + 3] = pack_bf16(y.z, y.w);
            }
        }
    }

    // ---- K loop: stage W in two 32KB halves, 64 WMMAs per wave ------------
    FragF32 acc[8];
#pragma unroll
    for (int n = 0; n < 8; ++n)
#pragma unroll
        for (int i = 0; i < 8; ++i) acc[n].f[i] = 0.f;

#pragma unroll
    for (int h = 0; h < 2; ++h) {
        __syncthreads();
        {
            const uint4* src = (const uint4*)(Wb + h * 8192);
            uint4* dst = (uint4*)ldsW;
            for (int i = tid; i < 2048; i += 256) dst[i] = src[i];
        }
        __syncthreads();
#pragma unroll
        for (int kl = 0; kl < 4; ++kl) {
            const int kk = h * 4 + kl;
#pragma unroll
            for (int n = 0; n < 8; ++n) {
                FragB16 bf;
                const uint4* p = (const uint4*)&ldsW[(((kl * 8 + n) * 32) + lane) * 8];
                uint4 w0 = p[0], w1 = p[1];
                bf.u[0] = w0.x; bf.u[1] = w0.y; bf.u[2] = w0.z; bf.u[3] = w0.w;
                bf.u[4] = w1.x; bf.u[5] = w1.y; bf.u[6] = w1.z; bf.u[7] = w1.w;
                acc[n].v = __builtin_amdgcn_wmma_f32_16x16x32_bf16(
                    false, afr[kk].v, false, bf.v, (short)0, acc[n].v, false, false);
            }
        }
    }

    // C/D layout: VGPR j / lanes<16 -> row M=j ; lanes>=16 -> row M=8+j ;
    // column = 16n + (lane&15). Valid rows are M<12: all j for lanes<16, j<4 hi.
    if (PASS == 1) {
#pragma unroll
        for (int n = 0; n < 8; ++n) {
            float s = 0.f, q = 0.f;
#pragma unroll
            for (int j = 0; j < 8; ++j) {
                float v = ((!hi) || (j < 4)) ? acc[n].f[j] : 0.f;
                s += v; q += v * v;
            }
            s += __shfl_xor(s, 16, 32);
            q += __shfl_xor(q, 16, 32);
            if (!hi) {
                int col = n * 16 + mrow;
                atomicAdd(&ldsStat[col], s);
                atomicAdd(&ldsStat[128 + col], q);
            }
        }
        __syncthreads();
        atomicAdd(&stats1[tid], ldsStat[tid]);
    } else {
        const float invR = 1.f / R1;
#pragma unroll
        for (int n = 0; n < 4; ++n) {
            int colF = n * 16 + mrow;   // filter column (0..63)
            int colC = colF + 64;       // core column   (64..127)
            float mF = stats1[colF] * invR;
            float vF = stats1[128 + colF] * invR - mF * mF;
            float gF = gamma1[colF] * rsqrtf(vF + EPSV);
            float bF = beta1[colF] - mF * gF;
            float mC = stats1[colC] * invR;
            float vC = stats1[128 + colC] * invR - mC * mC;
            float gC = gamma1[colC] * rsqrtf(vC + EPSV);
            float bC = beta1[colC] - mC * gC;
            float psum = 0.f;
#pragma unroll
            for (int j = 0; j < 8; ++j) {
                float yf = acc[n].f[j] * gF + bF;
                float yc = acc[n + 4].f[j] * gC + bC;
                float filt = 1.f / (1.f + __expf(-yf));
                float core = softplusf(yc);
                psum += ((!hi) || (j < 4)) ? filt * core : 0.f;
            }
            psum += __shfl_xor(psum, 16, 32);
            if (!hi && active) {
                nbr_sumed[(long)a * FA + colF] = psum;
                atomicAdd(&ldsStat[colF], psum);
                atomicAdd(&ldsStat[64 + colF], psum * psum);
            }
        }
        __syncthreads();
        if (tid < 128) atomicAdd(&stats2[tid], ldsStat[tid]);
    }
}

// ---------------------------------------------------------------------------
// Kernel 3: BN2 + residual + softplus, elementwise over [NA, 64].
// ---------------------------------------------------------------------------
__global__ __launch_bounds__(256) void conv_final(
    const float* __restrict__ atom, const float* __restrict__ nbr_sumed,
    const float* __restrict__ stats2, const float* __restrict__ gamma2,
    const float* __restrict__ beta2, float* __restrict__ out) {
    long i = (long)blockIdx.x * 256 + threadIdx.x;
    if (i >= (long)NA * FA) return;
    int c = (int)(i & 63);
    float mean = stats2[c] / R2;
    float var  = stats2[64 + c] / R2 - mean * mean;
    float bn = (nbr_sumed[i] - mean) * rsqrtf(var + EPSV) * gamma2[c] + beta2[c];
    out[i] = softplusf(atom[i] + bn);
}

// ---------------------------------------------------------------------------
extern "C" void kernel_launch(void* const* d_in, const int* in_sizes, int n_in,
                              void* d_out, int out_size, void* d_ws, size_t ws_size,
                              hipStream_t stream) {
    const float* atom   = (const float*)d_in[0];
    const float* nbrf   = (const float*)d_in[1];
    const int*   idx    = (const int*)d_in[2];
    const float* W      = (const float*)d_in[3];
    // d_in[4] = bias b: additive constant cancels exactly through BN1 -> unused
    const float* gamma1 = (const float*)d_in[5];
    const float* beta1  = (const float*)d_in[6];
    const float* gamma2 = (const float*)d_in[7];
    const float* beta2  = (const float*)d_in[8];

    float* ws      = (float*)d_ws;
    float* stats1  = ws;                         // 256 f32: sum1 | sumsq1
    float* stats2  = ws + 256;                   // 128 f32: sum2 | sumsq2
    unsigned* Wb   = (unsigned*)(ws + 512);      // 16384 words (64KB bf16 W)
    float* nbrs    = ws + 512 + 16384;           // NA*64 f32 nbr_sumed
    float* out     = (float*)d_out;

    conv_prep<<<64, 256, 0, stream>>>(W, Wb, stats1);

    const int gemm_blocks = (NA + 7) / 8;        // 12500
    conv_gemm<1><<<gemm_blocks, 256, 0, stream>>>(atom, nbrf, idx, Wb, gamma1,
                                                  beta1, stats1, stats2, nbrs);
    conv_gemm<2><<<gemm_blocks, 256, 0, stream>>>(atom, nbrf, idx, Wb, gamma1,
                                                  beta1, stats1, stats2, nbrs);

    const long total = (long)NA * FA;
    conv_final<<<(int)((total + 255) / 256), 256, 0, stream>>>(
        atom, nbrs, stats2, gamma2, beta2, out);
}
